// Cross_Att_5119601016905
// MI455X (gfx1250) — matrix operand order repaired and checked by reading, not compiled, not run on hardware
//
#include <hip/hip_runtime.h>
#include <hip/hip_bf16.h>

#define Bsz 64
#define Nn  512
#define Dd  128

typedef __bf16 bf16;
typedef __attribute__((ext_vector_type(16))) __bf16 v16bf;
typedef __attribute__((ext_vector_type(8)))  __bf16 v8bf;
typedef __attribute__((ext_vector_type(8)))  float  v8f;
typedef __attribute__((ext_vector_type(4)))  unsigned int v4u;
typedef __attribute__((ext_vector_type(8)))  int v8i;
typedef __attribute__((ext_vector_type(4)))  int v4i;

static __device__ __forceinline__ v16bf cat8(v8bf lo, v8bf hi) {
  return __builtin_shufflevector(lo, hi, 0,1,2,3,4,5,6,7,8,9,10,11,12,13,14,15);
}
static __device__ __forceinline__ v8f wmma_bf16(v16bf a, v16bf b, v8f c) {
  return __builtin_amdgcn_wmma_f32_16x16x32_bf16(false, a, false, b, (short)0, c,
                                                 false, false);
}

// ---------------------------------------------------------------------------
// Projection: k = h @ Wk^T + bk ; q = h @ Wq^T + bq ; output bf16.
// One wave = one 16-row tile over concat(h1,h2). 8 waves / block.
// ---------------------------------------------------------------------------
__global__ __launch_bounds__(256) void proj_kernel(
    const float* __restrict__ h1, const float* __restrict__ h2,
    const float* __restrict__ Wk, const float* __restrict__ bk,
    const float* __restrict__ Wq, const float* __restrict__ bq,
    bf16* __restrict__ k1, bf16* __restrict__ k2,
    bf16* __restrict__ q1, bf16* __restrict__ q2)
{
  const int WSTR = 136;                 // padded row stride (bf16 elems)
  __shared__ bf16 Wlds[128 * 136];

  const int tid  = threadIdx.x;
  const int wave = tid >> 5;
  const int lane = tid & 31;
  const int lh   = lane >> 4;           // half-wave id (0/1)
  const int lm   = lane & 15;

  const int tileId  = blockIdx.x * 8 + wave;        // [0, 4096)
  const int rowBase = tileId * 16;                  // row over concat(h1,h2)
  const bool src2   = rowBase >= Bsz * Nn;
  const float* hsrc = src2 ? h2 : h1;
  const int rloc    = rowBase - (src2 ? Bsz * Nn : 0);

  // A-fragments: 16 rows x 128 cols of h, bf16, 4 K-chunks of 32.
  v16bf afrag[4];
  {
    const float* hrow = hsrc + (size_t)(rloc + lm) * Dd;
#pragma unroll
    for (int c = 0; c < 4; ++c) {
      const float* plo = hrow + 32 * c + 8 * lh;
      const float* phi = plo + 16;
      float4 a0 = *(const float4*)plo;
      float4 a1 = *(const float4*)(plo + 4);
      float4 b0 = *(const float4*)phi;
      float4 b1 = *(const float4*)(phi + 4);
      v16bf f;
      f[0]=(bf16)a0.x; f[1]=(bf16)a0.y; f[2]=(bf16)a0.z; f[3]=(bf16)a0.w;
      f[4]=(bf16)a1.x; f[5]=(bf16)a1.y; f[6]=(bf16)a1.z; f[7]=(bf16)a1.w;
      f[8]=(bf16)b0.x; f[9]=(bf16)b0.y; f[10]=(bf16)b0.z; f[11]=(bf16)b0.w;
      f[12]=(bf16)b1.x; f[13]=(bf16)b1.y; f[14]=(bf16)b1.z; f[15]=(bf16)b1.w;
      afrag[c] = f;
    }
  }

  bf16* kdst = (src2 ? k2 : k1) + (size_t)rloc * Dd;
  bf16* qdst = (src2 ? q2 : q1) + (size_t)rloc * Dd;

  for (int pass = 0; pass < 2; ++pass) {
    const float* W    = pass ? Wq : Wk;
    const float* bias = pass ? bq : bk;
    bf16* dst         = pass ? qdst : kdst;

    __syncthreads();
    for (int i = tid; i < 128 * 16; i += 256) {     // f32 -> bf16 weight stage
      int e = i >> 4, d0 = (i & 15) << 3;
      float4 a = *(const float4*)(W + e * 128 + d0);
      float4 b = *(const float4*)(W + e * 128 + d0 + 4);
      v8bf v;
      v[0]=(bf16)a.x; v[1]=(bf16)a.y; v[2]=(bf16)a.z; v[3]=(bf16)a.w;
      v[4]=(bf16)b.x; v[5]=(bf16)b.y; v[6]=(bf16)b.z; v[7]=(bf16)b.w;
      *(v8bf*)&Wlds[e * WSTR + d0] = v;
    }
    __syncthreads();

#pragma unroll
    for (int et = 0; et < 8; ++et) {
      v8f acc = {};
#pragma unroll
      for (int c = 0; c < 4; ++c) {
        const bf16* wp = &Wlds[(16 * et + lm) * WSTR + 32 * c + 8 * lh];
        v8bf lo = *(const v8bf*)wp;
        v8bf hi = *(const v8bf*)(wp + 16);
        acc = wmma_bf16(afrag[c], cat8(lo, hi), acc);
      }
      float bv = bias[16 * et + lm];
#pragma unroll
      for (int v = 0; v < 8; ++v) {
        float o = acc[v] + bv;
        dst[(size_t)(v + 8 * lh) * Dd + 16 * et + lm] = (bf16)o;
      }
    }
  }
}

// ---------------------------------------------------------------------------
// Flash-style cross attention with TDM double-buffered K staging.
// One block = (batch, 128-query block, dir); 8 waves x 16 query rows.
// ---------------------------------------------------------------------------
__global__ __launch_bounds__(256) void attn_kernel(
    const bf16* __restrict__ q1, const bf16* __restrict__ q2,
    const bf16* __restrict__ k1, const bf16* __restrict__ k2,
    const int* __restrict__ len1, const int* __restrict__ len2,
    const float* __restrict__ tptr,
    float* __restrict__ out)
{
  const int KSTR = 136;   // Kbuf row stride (bf16) = 256B data + 16B pad
  const int TSTR = 40;    // Kt row stride
  const int PSTR = 40;    // Pbuf row stride
  __shared__ bf16 Kbuf[2][32 * 136];    // [key][d], TDM-filled, 2 x 8.5 KB
  __shared__ bf16 Ktb[2][128 * 40];     // [d][key] transposed, 2 x 10 KB
  __shared__ bf16 Pbuf[8][16 * 40];     // per-wave P stash, 10 KB

  const int tid  = threadIdx.x;
  const int wave = tid >> 5;
  const int lane = tid & 31;
  const int lh   = lane >> 4;
  const int lm   = lane & 15;

  const int b   = blockIdx.x >> 2;
  const int qb  = blockIdx.x & 3;
  const int dir = blockIdx.y;

  const bf16* Q  = (dir ? q2 : q1) + (size_t)b * Nn * Dd;
  const bf16* K  = (dir ? k1 : k2) + (size_t)b * Nn * Dd;
  const int lenQ = (dir ? len2 : len1)[b];
  const int lenK = (dir ? len1 : len2)[b];
  float* O       = out + (size_t)dir * Bsz * Nn * Dd + (size_t)b * Nn * Dd;
  const float tv = tptr[0];

  const int n0 = qb * 128 + wave * 16;

  // --- TDM issue: 2D tile 128(d) x 32(keys), bf16, padded rows (KSTR=136) ---
  auto tdm_issue = [&](int blk, int buf) {
    unsigned lds = (unsigned)(size_t)&Kbuf[buf][0];
    unsigned long long ga =
        (unsigned long long)(size_t)(K + (size_t)blk * 32 * Dd);
    v4u g0;
    g0[0] = 1u;                                    // count=1 valid descriptor
    g0[1] = lds;                                   // lds_addr (bytes)
    g0[2] = (unsigned)ga;                          // global_addr[31:0]
    g0[3] = (unsigned)((ga >> 32) & 0x01FFFFFFu)   // global_addr[56:32]
            | 0x80000000u;                         // type = 2 ("image")
    v8i g1;
    g1[0] = (1 << 16)      // data_size = 2 bytes
          | (1 << 20)      // pad_enable
          | (5 << 22)      // pad_interval: 64 DWORDs = 256 B
          | (3 << 25);     // pad_amount:   4 DWORDs =  16 B
    g1[1] = (128 << 16);   // tensor_dim0 = 128 (low 16 bits in [31:16])
    g1[2] = (32 << 16);    // tensor_dim0 hi=0 | tensor_dim1 = 32
    g1[3] = (128 << 16);   // tensor_dim1 hi=0 | tile_dim0 = 128
    g1[4] = 32;            // tile_dim1 = 32 | tile_dim2 = 0
    g1[5] = 128;           // tensor_dim0_stride = 128
    g1[6] = 0;             // stride0 hi | tensor_dim1_stride lo
    g1[7] = 0;
    v4i g2; g2[0] = 1; g2[1] = 0; g2[2] = 0; g2[3] = 0;   // tensor_dim2 = 1
    v4i g3; g3[0] = 0; g3[1] = 0; g3[2] = 0; g3[3] = 0;
    v8i g4; g4[0]=0; g4[1]=0; g4[2]=0; g4[3]=0; g4[4]=0; g4[5]=0; g4[6]=0; g4[7]=0;
    __builtin_amdgcn_tensor_load_to_lds(g0, g1, g2, g3, g4, 0);
  };

  // --- global loads feeding the transposed Ktb copy (2 x 8 bf16 / thread) ---
  const int ci0 = tid, ci1 = tid + 256;                  // chunk ids [0,512)
  const int key0 = ci0 >> 4, dA = (ci0 & 15) << 3;
  const int key1 = ci1 >> 4, dB = (ci1 & 15) << 3;
  auto gload = [&](int blk, v8bf& r0, v8bf& r1) {
    r0 = *(const v8bf*)(K + (size_t)(blk * 32 + key0) * Dd + dA);
    r1 = *(const v8bf*)(K + (size_t)(blk * 32 + key1) * Dd + dB);
  };
  auto scatter = [&](int buf, v8bf r0, v8bf r1) {
#pragma unroll
    for (int e = 0; e < 8; ++e) Ktb[buf][(dA + e) * TSTR + key0] = r0[e];
#pragma unroll
    for (int e = 0; e < 8; ++e) Ktb[buf][(dB + e) * TSTR + key1] = r1[e];
  };

  // Q A-fragments (16 x 128 bf16, 4 chunks)
  v16bf qf[4];
  {
    const bf16* qrow = Q + (size_t)(n0 + lm) * Dd;
#pragma unroll
    for (int c = 0; c < 4; ++c) {
      const bf16* p = qrow + 32 * c + 8 * lh;
      v8bf lo = *(const v8bf*)p;
      v8bf hi = *(const v8bf*)(p + 16);
      qf[c] = cat8(lo, hi);
    }
  }

  v8f Oacc[8];
  v8f zero = {};
#pragma unroll
  for (int dt = 0; dt < 8; ++dt) Oacc[dt] = zero;
  float mrow[8], lrow[8];
#pragma unroll
  for (int v = 0; v < 8; ++v) { mrow[v] = -3.0e38f; lrow[v] = 0.0f; }

  // ---- prologue: stage block 0 into buffer 0 ----
  v8bf r0, r1;
  if (wave == 0) tdm_issue(0, 0);
  gload(0, r0, r1);
  scatter(0, r0, r1);
  if (wave == 0) __builtin_amdgcn_s_wait_tensorcnt(0);
  __syncthreads();

  for (int blk = 0; blk < 16; ++blk) {
    const int cur = blk & 1;

    // kick off next block's staging before computing on the current one
    if (blk < 15) {
      if (wave == 0) tdm_issue(blk + 1, 1 - cur);
      gload(blk + 1, r0, r1);
    }

    // scores S = (Q K^T) * t  — two 16-key tiles, masked by lenK
    v8f S[2];
#pragma unroll
    for (int t2 = 0; t2 < 2; ++t2) {
      v8f acc = {};
#pragma unroll
      for (int c = 0; c < 4; ++c) {
        const bf16* p = &Kbuf[cur][(t2 * 16 + lm) * KSTR + 32 * c + 8 * lh];
        v8bf lo = *(const v8bf*)p;
        v8bf hi = *(const v8bf*)(p + 16);
        acc = wmma_bf16(qf[c], cat8(lo, hi), acc);
      }
      const int keyg = blk * 32 + t2 * 16 + lm;
      const bool kok = keyg < lenK;
#pragma unroll
      for (int v = 0; v < 8; ++v)
        S[t2][v] = kok ? acc[v] * tv : -1.0e9f;
    }

    // online softmax: row stats per VGPR (row = v + 8h), 16-lane reductions
#pragma unroll
    for (int v = 0; v < 8; ++v) {
      float tm = fmaxf(S[0][v], S[1][v]);
#pragma unroll
      for (int mm = 1; mm < 16; mm <<= 1)
        tm = fmaxf(tm, __shfl_xor(tm, mm, 32));
      float nm    = fmaxf(mrow[v], tm);
      float alpha = __expf(mrow[v] - nm);
      mrow[v]     = nm;
      float p0 = __expf(S[0][v] - nm);
      float p1 = __expf(S[1][v] - nm);
      float rs = p0 + p1;
#pragma unroll
      for (int mm = 1; mm < 16; mm <<= 1)
        rs += __shfl_xor(rs, mm, 32);
      lrow[v] = lrow[v] * alpha + rs;
#pragma unroll
      for (int dt = 0; dt < 8; ++dt) Oacc[dt][v] *= alpha;
      Pbuf[wave][(v + 8 * lh) * PSTR + lm]      = (bf16)p0;
      Pbuf[wave][(v + 8 * lh) * PSTR + 16 + lm] = (bf16)p1;
    }

    // reload P as A-fragment (16x32) and accumulate O += P @ V
    v16bf pf;
    {
      const bf16* p = &Pbuf[wave][lm * PSTR + 8 * lh];
      v8bf lo = *(const v8bf*)p;
      v8bf hi = *(const v8bf*)(p + 16);
      pf = cat8(lo, hi);
    }
#pragma unroll
    for (int dt = 0; dt < 8; ++dt) {
      const bf16* p = &Ktb[cur][(16 * dt + lm) * TSTR + 8 * lh];
      v8bf lo = *(const v8bf*)p;
      v8bf hi = *(const v8bf*)(p + 16);
      Oacc[dt] = wmma_bf16(pf, cat8(lo, hi), Oacc[dt]);
    }

    // finish staging the next block, then hand off
    if (blk < 15) {
      scatter(1 - cur, r0, r1);
      if (wave == 0) __builtin_amdgcn_s_wait_tensorcnt(0);
      __syncthreads();
    }
  }

  // epilogue: divide by row sums, zero invalid query rows, store f32
#pragma unroll
  for (int v = 0; v < 8; ++v) {
    const int row = n0 + v + 8 * lh;
    const float inv = (row < lenQ) ? 1.0f / lrow[v] : 0.0f;
    float* orow = O + (size_t)row * Dd;
#pragma unroll
    for (int dt = 0; dt < 8; ++dt)
      orow[16 * dt + lm] = Oacc[dt][v] * inv;
  }
}

// ---------------------------------------------------------------------------
extern "C" void kernel_launch(void* const* d_in, const int* in_sizes, int n_in,
                              void* d_out, int out_size, void* d_ws, size_t ws_size,
                              hipStream_t stream)
{
  const float* h1 = (const float*)d_in[0];
  const float* h2 = (const float*)d_in[1];
  const float* Wk = (const float*)d_in[2];
  const float* bk = (const float*)d_in[3];
  const float* Wq = (const float*)d_in[4];
  const float* bq = (const float*)d_in[5];
  const float* t  = (const float*)d_in[6];
  const int* len1 = (const int*)d_in[7];
  const int* len2 = (const int*)d_in[8];

  const size_t mat = (size_t)Bsz * Nn * Dd;       // 4M elems per matrix
  bf16* k1 = (bf16*)d_ws;                         // 4 bf16 matrices = 32 MB
  bf16* k2 = k1 + mat;
  bf16* q1 = k2 + mat;
  bf16* q2 = q1 + mat;

  proj_kernel<<<512, 256, 0, stream>>>(h1, h2, Wk, bk, Wq, bq, k1, k2, q1, q2);
  attn_kernel<<<dim3(Bsz * 4, 2), 256, 0, stream>>>(q1, q2, k1, k2,
                                                    len1, len2, t,
                                                    (float*)d_out);
}
